// PretrainGAT_47029891891569
// MI455X (gfx1250) — matrix operand order repaired and checked by reading, not compiled, not run on hardware
//
#include <hip/hip_runtime.h>
#include <hip/hip_bf16.h>

// ---------------------------------------------------------------------------
// CDNA5 (gfx1250) wave32 WMMA implementation of the PretrainGAT block.
// All heavy contractions (GEMMs + 3x3 convs as implicit GEMM) run on
// v_wmma_f32_16x16x32_bf16 with f32 accumulation; norm/reduction glue is f32.
// Conv = implicit GEMM: 1 workgroup stages a 3-row halo window in LDS once,
// then 8 waves (conv3) each produce 16x64 of output with zero further barriers.
// ---------------------------------------------------------------------------

typedef __attribute__((ext_vector_type(16))) __bf16 v16bf;
typedef __attribute__((ext_vector_type(8)))  __bf16 v8bf;
typedef __attribute__((ext_vector_type(8)))  float  v8f;

union ABfrag { v16bf v; v8bf h[2]; __bf16 e[16]; };

#define LDIM 512
#define NPIX 262144   // 512*512
#define MT   64       // pixels per conv workgroup (one image row segment)

// ---------------- elementwise helpers ----------------
__global__ void k_cast_bf16(const float* __restrict__ s, __bf16* __restrict__ d, long n) {
  long i = (long)blockIdx.x * 256 + threadIdx.x;
  if (i < n) d[i] = (__bf16)s[i];
}

__global__ void k_zero_bf16(__bf16* __restrict__ d, long n) {
  long i = (long)blockIdx.x * 256 + threadIdx.x;
  if (i < n) d[i] = (__bf16)0.f;
}

// repack OIHW conv weights -> [O][tap=9][Cpad] bf16, zero-padded channels
__global__ void k_pack_cw(const float* __restrict__ w, int I,
                          __bf16* __restrict__ bw, int Cpad, long n) {
  long i = (long)blockIdx.x * 256 + threadIdx.x;
  if (i >= n) return;
  int  c   = (int)(i % Cpad);
  long r   = i / Cpad;
  int  tap = (int)(r % 9);
  long o   = r / 9;
  bw[i] = (c < I) ? (__bf16)w[(o * I + c) * 9 + tap] : (__bf16)0.f;
}

// InstanceNorm2d stats: per-channel mean & rstd over all 512*512 pixels.
__global__ __launch_bounds__(256)
void k_inst_stats(const float* __restrict__ src, int C,
                  float* __restrict__ mean, float* __restrict__ rstd) {
  int c = blockIdx.x;
  float s = 0.f, ss = 0.f;
  for (long i = threadIdx.x; i < NPIX; i += 256) {
    float v = src[i * C + c];
    s += v; ss += v * v;
  }
  __shared__ float sh[256], sh2[256];
  sh[threadIdx.x] = s; sh2[threadIdx.x] = ss;
  __syncthreads();
  for (int o = 128; o > 0; o >>= 1) {
    if (threadIdx.x < (unsigned)o) {
      sh[threadIdx.x]  += sh[threadIdx.x + o];
      sh2[threadIdx.x] += sh2[threadIdx.x + o];
    }
    __syncthreads();
  }
  if (threadIdx.x == 0) {
    float m = sh[0] / (float)NPIX;
    float v = sh2[0] / (float)NPIX - m * m;
    mean[c] = m;
    rstd[c] = rsqrtf(v + 1e-5f);
  }
}

// normalize + cast + scatter into padded NHWC bf16 conv input
__global__ void k_pack_norm(const float* __restrict__ src, int C,
                            const float* __restrict__ mean, const float* __restrict__ rstd,
                            __bf16* __restrict__ dst, int Cpad, int coff, long n) {
  long i = (long)blockIdx.x * 256 + threadIdx.x;
  if (i >= n) return;
  long pix = i / C;
  int  c   = (int)(i % C);
  dst[pix * Cpad + coff + c] = (__bf16)((src[i] - mean[c]) * rstd[c]);
}

// alpha (pix,16) f32  ->  aT[h][pix] bf16 (head-major WMMA A operand)
__global__ void k_transpose_alpha(const float* __restrict__ a, __bf16* __restrict__ aT) {
  long i = (long)blockIdx.x * 256 + threadIdx.x;
  if (i >= (long)NPIX * 16) return;
  long pix = i >> 4;
  int  h   = (int)(i & 15);
  aT[(long)h * NPIX + pix] = (__bf16)a[i];
}

// v (512x512 bf16, row j, col c=h*32+d) -> vT[c][j]  (contiguous-K B operand)
__global__ void k_transpose_v(const __bf16* __restrict__ v, __bf16* __restrict__ vT) {
  long i = (long)blockIdx.x * 256 + threadIdx.x;
  if (i >= (long)NPIX) return;
  long r = i >> 9, c = i & 511;
  vT[c * 512 + r] = v[i];
}

// LayerNorm over last dim; optional residual add, optional leaky(0.01),
// output to f32 and/or bf16.
__global__ __launch_bounds__(256)
void k_ln(const float* __restrict__ src, const float* __restrict__ add, int C,
          const float* __restrict__ g, const float* __restrict__ b,
          float* __restrict__ outf, __bf16* __restrict__ outb, int leaky) {
  long r = blockIdx.x;
  const float* p  = src + r * C;
  const float* p2 = add ? add + r * C : nullptr;
  float s = 0.f, ss = 0.f;
  for (int i = threadIdx.x; i < C; i += 256) {
    float v = p[i] + (p2 ? p2[i] : 0.f);
    s += v; ss += v * v;
  }
  __shared__ float sh[256], sh2[256];
  sh[threadIdx.x] = s; sh2[threadIdx.x] = ss;
  __syncthreads();
  for (int o = 128; o > 0; o >>= 1) {
    if (threadIdx.x < (unsigned)o) {
      sh[threadIdx.x]  += sh[threadIdx.x + o];
      sh2[threadIdx.x] += sh2[threadIdx.x + o];
    }
    __syncthreads();
  }
  __shared__ float mS, rS;
  if (threadIdx.x == 0) {
    float m = sh[0] / (float)C;
    float v = sh2[0] / (float)C - m * m;
    mS = m; rS = rsqrtf(v + 1e-5f);
  }
  __syncthreads();
  for (int i = threadIdx.x; i < C; i += 256) {
    float v = p[i] + (p2 ? p2[i] : 0.f);
    v = (v - mS) * rS * g[i] + b[i];
    if (leaky && v < 0.f) v *= 0.01f;
    if (outf) outf[r * C + i] = v;
    if (outb) outb[r * C + i] = (__bf16)v;
  }
}

// ---------------------------------------------------------------------------
// Generic batched WMMA GEMM, one wave per 16x16 C tile.
//   BT==1: C[m][n] = scale * sum_k A[m][k] * B[n][k]   (B stored [N][K])
//   BT==0: C[m][n] = scale * sum_k A[m][k] * B[k][n]   (B stored [K][N])
// Fragments use contiguous k order (lane half h -> K 16h..16h+15) for A and B alike.
// ---------------------------------------------------------------------------
template <int BT>
__global__ __launch_bounds__(32)
void k_gemm(const __bf16* __restrict__ A, long lda, long bsA,
            const __bf16* __restrict__ B, long ldb, long bsB,
            float* __restrict__ Cf, __bf16* __restrict__ Cb,
            long ldc, long cstride, long bsC,
            const float* __restrict__ bias, float scale, int K) {
  int lane = threadIdx.x;
  int m    = lane & 15;
  int half = lane >> 4;
  int n    = m;                       // B-column owned by this lane
  long m0  = (long)blockIdx.x * 16;
  long n0  = (long)blockIdx.y * 16;
  long bz  = blockIdx.z;

  const __bf16* Ab = A + bz * bsA + (m0 + m) * lda;
  v8f acc = {};
  for (int k0 = 0; k0 < K; k0 += 32) {
    ABfrag a, b;
    // A 16x32: lane half 0 -> K {0..15}; half 1 -> K {16..31}
    a.h[0] = *(const v8bf*)(Ab + k0 + half * 16);
    a.h[1] = *(const v8bf*)(Ab + k0 + half * 16 + 8);
    if (BT) {
      const __bf16* br = B + bz * bsB + (n0 + n) * ldb + k0 + half * 16;
      b.h[0] = *(const v8bf*)br;
      b.h[1] = *(const v8bf*)(br + 8);
    } else {
      const __bf16* bc = B + bz * bsB + (long)(k0 + half * 16) * ldb + (n0 + n);
#pragma unroll
      for (int i2 = 0; i2 < 16; ++i2) b.e[i2] = bc[(long)i2 * ldb];
    }
    acc = __builtin_amdgcn_wmma_f32_16x16x32_bf16(false, a.v, false, b.v,
                                                  (short)0, acc, false, false);
  }
  float bv = bias ? bias[n0 + n] : 0.f;
#pragma unroll
  for (int j = 0; j < 8; ++j) {
    long row = m0 + half * 8 + j;     // C row for acc[j]
    float v  = acc[j] * scale + bv;
    long idx = bz * bsC + row * ldc + (n0 + n) * cstride;
    if (Cf) Cf[idx] = v;
    if (Cb) Cb[idx] = (__bf16)v;
  }
}

// ---------------------------------------------------------------------------
// Implicit-GEMM 3x3 SAME conv (stride 1) over 512x512, NHWC bf16 input,
// channels padded to Cpad (mult. of 32), weights [Cout][9][Cpad] bf16.
// Workgroup = (Cout/16) waves, covers MT=64 pixels (one row segment) x Cout.
// LDS stages the 3-row halo window (3 x 66 x Cpad bf16) ONCE; compute loop is
// barrier-free: per (tap, chunk) each wave loads one B fragment and reuses it
// across 4 WMMAs (4 M-subtiles). Epilogue: +bias, LeakyReLU(0.01), f32 NHWC.
// ---------------------------------------------------------------------------
__global__ __launch_bounds__(256)
void k_conv3x3(const __bf16* __restrict__ pin, int Cpad,
               const __bf16* __restrict__ Bw, const float* __restrict__ bias,
               int Cout, float* __restrict__ out) {
  extern __shared__ __align__(16) __bf16 Ls[];   // [3][MT+2][Cpad]
  const int tid  = threadIdx.x;
  const int lane = tid & 31;
  const int wv   = tid >> 5;          // wave id -> cout tile
  const int m    = lane & 15;
  const int half = lane >> 4;
  const int n    = m;
  const long pix0 = (long)blockIdx.x * MT;
  const int  y    = (int)(pix0 >> 9);
  const int  x0   = (int)(pix0 & 511);
  const long n0   = (long)wv * 16;
  const int  cv8  = Cpad >> 3;        // channel-vectors (v8bf) per pixel

  // ---- stage halo window: rows y-1..y+1, pixels x0-1..x0+64 ----
  const int nvec = 3 * (MT + 2) * cv8;
  for (int t = tid; t < nvec; t += blockDim.x) {
    int cv  = t % cv8;
    int rp  = t / cv8;
    int pix = rp % (MT + 2);
    int row = rp / (MT + 2);
    int yy  = y + row - 1;
    int xx  = x0 + pix - 1;
    v8bf val = {};
    if (yy >= 0 && yy < LDIM && xx >= 0 && xx < LDIM)
      val = *(const v8bf*)(pin + ((long)yy * LDIM + xx) * Cpad + cv * 8);
    *(v8bf*)&Ls[(long)rp * Cpad + cv * 8] = val;
  }
  __syncthreads();

  // ---- barrier-free WMMA loop ----
  v8f acc[4] = {};
  for (int tap = 0; tap < 9; ++tap) {
    const int row = tap / 3;          // staged row (dy + 1)
    const int px  = tap % 3;          // dx + 1
    for (int cb = 0; cb < Cpad; cb += 32) {
      ABfrag b;
      const __bf16* br = Bw + ((n0 + n) * 9 + tap) * Cpad + cb + half * 16;
      b.h[0] = *(const v8bf*)br;
      b.h[1] = *(const v8bf*)(br + 8);
#pragma unroll
      for (int ms = 0; ms < 4; ++ms) {
        ABfrag a;
        const __bf16* ar = &Ls[((long)row * (MT + 2) + ms * 16 + m + px) * Cpad + cb];
        a.h[0] = *(const v8bf*)(ar + half * 16);
        a.h[1] = *(const v8bf*)(ar + half * 16 + 8);
        acc[ms] = __builtin_amdgcn_wmma_f32_16x16x32_bf16(false, a.v, false, b.v,
                                                          (short)0, acc[ms], false, false);
      }
    }
  }

  // ---- epilogue ----
  const float bv = bias[n0 + n];
#pragma unroll
  for (int ms = 0; ms < 4; ++ms) {
#pragma unroll
    for (int j = 0; j < 8; ++j) {
      long pixel = pix0 + ms * 16 + half * 8 + j;
      float v = acc[ms][j] + bv;
      v = (v < 0.f) ? 0.01f * v : v;
      out[pixel * Cout + n0 + n] = v;
    }
  }
}

// ---------------------------------------------------------------------------
extern "C" void kernel_launch(void* const* d_in, const int* in_sizes, int n_in,
                              void* d_out, int out_size, void* d_ws, size_t ws_size,
                              hipStream_t stream) {
  const float* x     = (const float*)d_in[0];
  const float* z     = (const float*)d_in[1];
  const float* plddt = (const float*)d_in[2];
  const float* Wq    = (const float*)d_in[3];
  const float* Wk    = (const float*)d_in[4];
  const float* Wv    = (const float*)d_in[5];
  const float* Wp    = (const float*)d_in[6];
  const float* cw1   = (const float*)d_in[7];
  const float* cb1   = (const float*)d_in[8];
  const float* cw2   = (const float*)d_in[9];
  const float* cb2   = (const float*)d_in[10];
  const float* ln1_g = (const float*)d_in[11];
  const float* ln1_b = (const float*)d_in[12];
  const float* W1    = (const float*)d_in[13];
  const float* b1    = (const float*)d_in[14];
  const float* ln2_g = (const float*)d_in[15];
  const float* ln2_b = (const float*)d_in[16];
  const float* W2    = (const float*)d_in[17];
  const float* b2    = (const float*)d_in[18];
  const float* lno_g = (const float*)d_in[19];
  const float* lno_b = (const float*)d_in[20];
  const float* cw3   = (const float*)d_in[21];
  const float* cb3   = (const float*)d_in[22];
  float* out = (float*)d_out;            // [x_new (262144) | pair_out (33554432)]

  // ---- workspace layout ----
  char* base = (char*)d_ws;
  size_t off = 0;
  auto alloc = [&](size_t bytes) -> void* {
    void* p = base + off;
    off = (off + bytes + 255) & ~(size_t)255;
    return p;
  };
  __bf16* xb    = (__bf16*)alloc((size_t)NPIX * 2);
  __bf16* wqb   = (__bf16*)alloc((size_t)NPIX * 2);
  __bf16* wkb   = (__bf16*)alloc((size_t)NPIX * 2);
  __bf16* wvb   = (__bf16*)alloc((size_t)NPIX * 2);
  __bf16* wpb   = (__bf16*)alloc(2048 * 2);
  __bf16* w1b   = (__bf16*)alloc((size_t)2560 * 1024 * 2);
  __bf16* w2b   = (__bf16*)alloc((size_t)1024 * 512 * 2);
  __bf16* zb    = (__bf16*)alloc((size_t)NPIX * 128 * 2);
  __bf16* qb    = (__bf16*)alloc((size_t)NPIX * 2);
  __bf16* kb    = (__bf16*)alloc((size_t)NPIX * 2);
  __bf16* vb    = (__bf16*)alloc((size_t)NPIX * 2);
  __bf16* vTb   = (__bf16*)alloc((size_t)NPIX * 2);
  float*  ap    = (float*)alloc((size_t)NPIX * 16 * 4);
  float*  an    = (float*)alloc((size_t)NPIX * 16 * 4);
  __bf16* pin12 = (__bf16*)alloc((size_t)NPIX * 32 * 2);
  __bf16* bw1   = (__bf16*)alloc((size_t)16 * 9 * 32 * 2);
  __bf16* bw2   = (__bf16*)alloc((size_t)16 * 9 * 32 * 2);
  __bf16* bw3   = (__bf16*)alloc((size_t)128 * 9 * 160 * 2);
  float*  a1    = (float*)alloc((size_t)NPIX * 16 * 4);
  float*  alpha2= (float*)alloc((size_t)NPIX * 16 * 4);
  __bf16* aT    = (__bf16*)alloc((size_t)16 * NPIX * 2);
  float*  hbuf  = (float*)alloc((size_t)512 * 2560 * 4);
  __bf16* hn1   = (__bf16*)alloc((size_t)512 * 2560 * 2);
  float*  h1    = (float*)alloc((size_t)512 * 1024 * 4);
  __bf16* hn2   = (__bf16*)alloc((size_t)512 * 1024 * 2);
  float*  xout  = (float*)alloc((size_t)512 * 512 * 4);
  float*  m1    = (float*)alloc(32 * 4);
  float*  r1    = (float*)alloc(32 * 4);
  float*  m2    = (float*)alloc(32 * 4);
  float*  r2    = (float*)alloc(32 * 4);
  float*  m3    = (float*)alloc(160 * 4);
  float*  r3    = (float*)alloc(160 * 4);
  __bf16* pin3  = (__bf16*)alloc((size_t)NPIX * 160 * 2);
  (void)ws_size; (void)in_sizes; (void)n_in; (void)out_size;

  auto cdiv = [](long a, long b) { return (a + b - 1) / b; };

  // ---- casts / repacks ----
  k_cast_bf16<<<cdiv(NPIX,256),256,0,stream>>>(x,  xb,  NPIX);
  k_cast_bf16<<<cdiv(NPIX,256),256,0,stream>>>(Wq, wqb, NPIX);
  k_cast_bf16<<<cdiv(NPIX,256),256,0,stream>>>(Wk, wkb, NPIX);
  k_cast_bf16<<<cdiv(NPIX,256),256,0,stream>>>(Wv, wvb, NPIX);
  k_cast_bf16<<<cdiv(2048,256),256,0,stream>>>(Wp, wpb, 2048);
  k_cast_bf16<<<cdiv((long)2560*1024,256),256,0,stream>>>(W1, w1b, (long)2560*1024);
  k_cast_bf16<<<cdiv((long)1024*512,256),256,0,stream>>>(W2, w2b, (long)1024*512);
  k_cast_bf16<<<cdiv((long)NPIX*128,256),256,0,stream>>>(z, zb, (long)NPIX*128);
  k_pack_cw<<<cdiv(16*9*32,256),256,0,stream>>>(cw1, 32, bw1, 32, 16*9*32);
  k_pack_cw<<<cdiv(16*9*32,256),256,0,stream>>>(cw2, 17, bw2, 32, 16*9*32);
  k_pack_cw<<<cdiv(128*9*160,256),256,0,stream>>>(cw3, 144, bw3, 160, 128*9*160);

  // ---- q,k,v = x @ W{q,k,v}.T  (bf16 out for downstream WMMA) ----
  k_gemm<1><<<dim3(32,32,1),32,0,stream>>>(xb,512,0, wqb,512,0, nullptr,qb, 512,1,0, nullptr,1.f,512);
  k_gemm<1><<<dim3(32,32,1),32,0,stream>>>(xb,512,0, wkb,512,0, nullptr,kb, 512,1,0, nullptr,1.f,512);
  k_gemm<1><<<dim3(32,32,1),32,0,stream>>>(xb,512,0, wvb,512,0, nullptr,vb, 512,1,0, nullptr,1.f,512);
  k_transpose_v<<<cdiv(NPIX,256),256,0,stream>>>(vb, vTb);

  // ---- alpha_pair = z @ Wp.T ; alpha_node = q.kT / sqrt(HD) per head ----
  k_gemm<1><<<dim3(NPIX/16,1,1),32,0,stream>>>(zb,128,0, wpb,128,0, ap,nullptr, 16,1,0, nullptr,1.f,128);
  k_gemm<1><<<dim3(32,32,16),32,0,stream>>>(qb,512,32, kb,512,32, an,nullptr, 8192,16,1,
                                            nullptr, 0.17677669529663687f, 32);

  // ---- conv1: inorm(concat[ap, an]) -> 3x3 (32->16) -> leaky ----
  k_inst_stats<<<16,256,0,stream>>>(ap, 16, m1,    r1);
  k_inst_stats<<<16,256,0,stream>>>(an, 16, m1+16, r1+16);
  k_zero_bf16<<<cdiv((long)NPIX*32,256),256,0,stream>>>(pin12, (long)NPIX*32);
  k_pack_norm<<<cdiv((long)NPIX*16,256),256,0,stream>>>(ap,16, m1,r1,       pin12,32,0,  (long)NPIX*16);
  k_pack_norm<<<cdiv((long)NPIX*16,256),256,0,stream>>>(an,16, m1+16,r1+16, pin12,32,16, (long)NPIX*16);
  k_conv3x3<<<dim3(NPIX/MT),32, 3*(MT+2)*32*2, stream>>>(pin12, 32, bw1, cb1, 16, a1);

  // ---- conv2: inorm(concat[a1, plddt]) -> 3x3 (17->16) -> leaky => alpha ----
  k_inst_stats<<<16,256,0,stream>>>(a1,    16, m2,    r2);
  k_inst_stats<<<1, 256,0,stream>>>(plddt,  1, m2+16, r2+16);
  k_zero_bf16<<<cdiv((long)NPIX*32,256),256,0,stream>>>(pin12, (long)NPIX*32);
  k_pack_norm<<<cdiv((long)NPIX*16,256),256,0,stream>>>(a1,16,   m2,r2,       pin12,32,0,  (long)NPIX*16);
  k_pack_norm<<<cdiv((long)NPIX,256),256,0,stream>>>(plddt,1,    m2+16,r2+16, pin12,32,16, (long)NPIX);
  k_conv3x3<<<dim3(NPIX/MT),32, 3*(MT+2)*32*2, stream>>>(pin12, 32, bw2, cb2, 16, alpha2);

  // ---- head-major bf16 alpha for aggregation GEMMs ----
  k_transpose_alpha<<<cdiv((long)NPIX*16,256),256,0,stream>>>(alpha2, aT);

  // node_from_pair: per i (batch=512): [16h x 512j] x [512j x 128p] -> h[:,0:2048]
  k_gemm<0><<<dim3(1,8,512),32,0,stream>>>(aT,(long)NPIX,512, zb,128,(long)512*128,
                                           hbuf,nullptr, 128,1,2560, nullptr,1.f,512);
  // node_from_node: per h (batch=16): [512i x 512j] x [32d x 512j]^T -> h[:,2048:2560]
  k_gemm<1><<<dim3(32,2,16),32,0,stream>>>(aT,512,(long)NPIX, vTb,512,(long)32*512,
                                           hbuf+2048,nullptr, 2560,1,32, nullptr,1.f,512);

  // ---- out_transform: LN -> Linear -> LN -> leaky -> Linear -> LN(res) ----
  k_ln<<<512,256,0,stream>>>(hbuf, nullptr, 2560, ln1_g, ln1_b, nullptr, hn1, 0);
  k_gemm<1><<<dim3(32,64,1),32,0,stream>>>(hn1,2560,0, w1b,2560,0, h1,nullptr, 1024,1,0, b1,1.f,2560);
  k_ln<<<512,256,0,stream>>>(h1, nullptr, 1024, ln2_g, ln2_b, nullptr, hn2, 1);
  k_gemm<1><<<dim3(32,32,1),32,0,stream>>>(hn2,1024,0, w2b,1024,0, xout,nullptr, 512,1,0, b2,1.f,1024);
  k_ln<<<512,256,0,stream>>>(xout, x, 512, lno_g, lno_b, out, nullptr, 0);   // x_new

  // ---- conv3: inorm(concat[z, alpha]) -> 3x3 (144->128) -> leaky => pair_out ----
  k_inst_stats<<<128,256,0,stream>>>(z,     128, m3,     r3);
  k_inst_stats<<<16, 256,0,stream>>>(alpha2, 16, m3+128, r3+128);
  k_zero_bf16<<<cdiv((long)NPIX*160,256),256,0,stream>>>(pin3, (long)NPIX*160);
  k_pack_norm<<<cdiv((long)NPIX*128,256),256,0,stream>>>(z,128,      m3,r3,         pin3,160,0,   (long)NPIX*128);
  k_pack_norm<<<cdiv((long)NPIX*16,256),256,0,stream>>>(alpha2,16,   m3+128,r3+128, pin3,160,128, (long)NPIX*16);
  k_conv3x3<<<dim3(NPIX/MT),256, 3*(MT+2)*160*2, stream>>>(pin3, 160, bw3, cb3, 128, out + NPIX);
}
